// EdgeConvLayer_45097156608062
// MI455X (gfx1250) — matrix operand order repaired and checked by reading, not compiled, not run on hardware
//
#include <hip/hip_runtime.h>

typedef __attribute__((ext_vector_type(16))) _Float16 v16h;
typedef __attribute__((ext_vector_type(8)))  _Float16 v8h;
typedef __attribute__((ext_vector_type(8)))  float    v8f;

#define N_NODES   100000
#define N_EDGES   1600000
#define IN_DIM    64
#define OUT_DIM   64
#define BN_EPS    1e-5f
#define LRELU     0.3f
#define NT_TILES  4            // 64 output cols / 16
#define KCHUNKS   2            // 64 K / 32
#define WGT_HALFS (2*NT_TILES*KCHUNKS*32*16)   // 8192 f16 = 16 KB
#define TILES     ((N_EDGES + 15) / 16)

#define HFEAT_BYTES ((size_t)N_NODES * IN_DIM * 2)           // 12.8 MB
#define WGT_BYTES   ((size_t)WGT_HALFS * 2)                  // 16 KB
#define STATS_BYTES ((size_t)256 * 4)                        // 1 KB
#define CNT_BYTES   ((size_t)N_NODES * 4)                    // 400 KB

// ---------------------------------------------------------------- WMMA core

__device__ __forceinline__ v8f wmma16(v16h a, v16h b, v8f c) {
  // D = A(16x32 f16) * B(32x16 f16) + C(16x16 f32)
  return __builtin_amdgcn_wmma_f32_16x16x32_f16(
      /*neg_a=*/false, a, /*neg_b=*/false, b,
      /*c_mod=*/(short)0, c, /*reuse_a=*/false, /*reuse_b=*/false);
}

// A-matrix 16x32 f16 layout (wave32):
//   lane L (L<16): row=L, elems 0..7 = K k0..k0+7, 8..15 = K k0+16..k0+23
//   lane L+16    : row=L, elems 0..7 = K k0+8..k0+15, 8..15 = K k0+24..k0+31

// fp32 feature gather (fallback path): 4x b128 + pack-convert
__device__ __forceinline__ v16h load_a_chunk(const float* __restrict__ frow,
                                             int kc, int hi, bool valid) {
  v16h a;
  if (valid) {
    const float4* p = (const float4*)frow;
    const int q = (kc * 32 + hi * 8) >> 2;
    float4 g0 = p[q + 0];
    float4 g1 = p[q + 1];
    float4 g2 = p[q + 4];                        // +16 floats
    float4 g3 = p[q + 5];
    a[0]  = (_Float16)g0.x; a[1]  = (_Float16)g0.y;
    a[2]  = (_Float16)g0.z; a[3]  = (_Float16)g0.w;
    a[4]  = (_Float16)g1.x; a[5]  = (_Float16)g1.y;
    a[6]  = (_Float16)g1.z; a[7]  = (_Float16)g1.w;
    a[8]  = (_Float16)g2.x; a[9]  = (_Float16)g2.y;
    a[10] = (_Float16)g2.z; a[11] = (_Float16)g2.w;
    a[12] = (_Float16)g3.x; a[13] = (_Float16)g3.y;
    a[14] = (_Float16)g3.z; a[15] = (_Float16)g3.w;
  } else {
    #pragma unroll
    for (int i = 0; i < 16; ++i) a[i] = (_Float16)0.0f;
  }
  return a;
}

// pre-packed f16 feature gather (fast path): 2x b128, zero VALU conversion
__device__ __forceinline__ v16h load_a_chunk(const unsigned short* __restrict__ frow,
                                             int kc, int hi, bool valid) {
  if (valid) {
    const int k0 = kc * 32 + hi * 8;             // in halfs; 16B aligned
    v8h g0 = *(const v8h*)(frow + k0);
    v8h g1 = *(const v8h*)(frow + k0 + 16);
    return __builtin_shufflevector(g0, g1,
        0, 1, 2, 3, 4, 5, 6, 7, 8, 9, 10, 11, 12, 13, 14, 15);
  }
  v16h z;
  #pragma unroll
  for (int i = 0; i < 16; ++i) z[i] = (_Float16)0.0f;
  return z;
}

// Computes m (pre-BN; linear bias cancels through BN) for one 16-edge tile.
// acc[nt]: C layout — lane holds (row = r + 8*(lane>=16), col = nt*16 + lane&15).
// Returns dst node index of row (lane & 15).
template <typename FT>
__device__ __forceinline__ int edge_tile_gemm(
    const FT* __restrict__ feature, const int* __restrict__ eidx,
    const _Float16* sW, int tile, int lane, v8f acc[NT_TILES]) {
  const int row  = lane & 15;
  const int hi   = lane >> 4;
  const int edge = tile * 16 + row;
  const bool valid = edge < N_EDGES;
  int sidx = 0, didx = 0;
  if (valid) {
    sidx = eidx[edge];             // src -> x_j
    didx = eidx[N_EDGES + edge];   // dst -> x_i
  }
  const FT* fi = feature + (size_t)didx * IN_DIM;
  const FT* fj = feature + (size_t)sidx * IN_DIM;

  const v8f z = {0.f,0.f,0.f,0.f,0.f,0.f,0.f,0.f};
  #pragma unroll
  for (int nt = 0; nt < NT_TILES; ++nt) acc[nt] = z;

  #pragma unroll
  for (int kc = 0; kc < KCHUNKS; ++kc) {
    v16h ai = load_a_chunk(fi, kc, hi, valid);   // x_i rows
    v16h aj = load_a_chunk(fj, kc, hi, valid);   // x_j rows
    #pragma unroll
    for (int nt = 0; nt < NT_TILES; ++nt) {
      v16h ba = *(const v16h*)(sW + (size_t)((((0*NT_TILES + nt)*KCHUNKS) + kc)*32 + lane)*16);
      v16h bb = *(const v16h*)(sW + (size_t)((((1*NT_TILES + nt)*KCHUNKS) + kc)*32 + lane)*16);
      acc[nt] = wmma16(ai, ba, acc[nt]);   // (W1-W2) @ x_i
      acc[nt] = wmma16(aj, bb, acc[nt]);   // W2 @ x_j
    }
  }
  return didx;
}

// ---------------------------------------------------------------- kernels

__global__ void zero_kernel(float* out, float* cnt, float* stats) {
  const size_t stride = (size_t)gridDim.x * blockDim.x;
  const size_t i0 = (size_t)blockIdx.x * blockDim.x + threadIdx.x;
  for (size_t i = i0; i < (size_t)N_NODES * OUT_DIM; i += stride) out[i] = 0.0f;
  for (size_t i = i0; i < (size_t)N_NODES; i += stride) cnt[i] = 0.0f;
  for (size_t i = i0; i < 256; i += stride) stats[i] = 0.0f;
}

// Pre-convert fp32 feature table to packed f16 (one-time, bandwidth bound).
__global__ void convert_features(const float* __restrict__ f,
                                 unsigned short* __restrict__ h) {
  const size_t i = (size_t)blockIdx.x * blockDim.x + threadIdx.x;
  if (i < (size_t)N_NODES * IN_DIM) {
    const _Float16 v = (_Float16)f[i];
    h[i] = *(const unsigned short*)&v;
  }
}

// Pack Wa = W[:, :64]-W[:, 64:], Wb = W[:, 64:] into WMMA B layout (f16).
// t bits: e[3:0] lane[8:4] kc[9] nt[11:10] mat[12]
__global__ void prep_weights(const float* __restrict__ W,
                             unsigned short* __restrict__ wgt) {
  const int t = blockIdx.x * blockDim.x + threadIdx.x;
  if (t >= WGT_HALFS) return;
  const int e    = t & 15;
  const int lane = (t >> 4) & 31;
  const int kc   = (t >> 9) & 1;
  const int nt   = (t >> 10) & 3;
  const int mat  = (t >> 12) & 1;
  const int n = nt * 16 + (lane & 15);
  const int k = kc * 32 + ((lane >> 4) & 1) * 8 + (e & 7) + (e >> 3) * 16;
  const float w1 = W[n * 128 + k];
  const float w2 = W[n * 128 + 64 + k];
  const float v = (mat == 0) ? (w1 - w2) : w2;
  const _Float16 hh = (_Float16)v;
  wgt[t] = *(const unsigned short*)&hh;
}

template <typename FT>
__global__ void __launch_bounds__(128) stage_stats(
    const FT* __restrict__ feature, const int* __restrict__ eidx,
    const unsigned short* __restrict__ wgt,
    float* __restrict__ gsum, float* __restrict__ gsq) {
  __shared__ __align__(16) _Float16 sW[WGT_HALFS];
  __shared__ float lsum[OUT_DIM], lsq[OUT_DIM];
  {
    const uint4* g = (const uint4*)wgt;
    uint4* s = (uint4*)sW;
    for (int i = threadIdx.x; i < WGT_HALFS / 8; i += blockDim.x) s[i] = g[i];
  }
  if (threadIdx.x < OUT_DIM) { lsum[threadIdx.x] = 0.f; lsq[threadIdx.x] = 0.f; }
  __syncthreads();

  const int lane   = threadIdx.x & 31;
  const int wave   = threadIdx.x >> 5;
  const int waveId = blockIdx.x * (blockDim.x >> 5) + wave;
  const int nWaves = gridDim.x * (blockDim.x >> 5);

  float rs[NT_TILES], rq[NT_TILES];
  #pragma unroll
  for (int nt = 0; nt < NT_TILES; ++nt) { rs[nt] = 0.f; rq[nt] = 0.f; }

  v8f acc[NT_TILES];
  for (int t = waveId; t < TILES; t += nWaves) {
    edge_tile_gemm(feature, eidx, sW, t, lane, acc);
    #pragma unroll
    for (int nt = 0; nt < NT_TILES; ++nt) {
      #pragma unroll
      for (int r = 0; r < 8; ++r) {
        const float v = acc[nt][r];
        rs[nt] += v;
        rq[nt] += v * v;
      }
    }
  }
  #pragma unroll
  for (int nt = 0; nt < NT_TILES; ++nt) {
    const int col = nt * 16 + (lane & 15);
    atomicAdd(&lsum[col], rs[nt]);
    atomicAdd(&lsq[col],  rq[nt]);
  }
  __syncthreads();
  if (threadIdx.x < OUT_DIM) {
    atomicAdd(&gsum[threadIdx.x], lsum[threadIdx.x]);
    atomicAdd(&gsq[threadIdx.x],  lsq[threadIdx.x]);
  }
}

__global__ void finalize_params(const float* __restrict__ gsum,
                                const float* __restrict__ gsq,
                                const float* __restrict__ gamma,
                                const float* __restrict__ beta,
                                float* __restrict__ pa, float* __restrict__ pc) {
  const int t = threadIdx.x;
  if (t < OUT_DIM) {
    const float mu  = gsum[t] * (1.0f / (float)N_EDGES);
    float var = gsq[t] * (1.0f / (float)N_EDGES) - mu * mu;
    var = fmaxf(var, 0.0f);
    const float a = gamma[t] * rsqrtf(var + BN_EPS);
    pa[t] = a;
    pc[t] = beta[t] - mu * a;   // linear bias b cancels through BN
  }
}

template <typename FT>
__global__ void __launch_bounds__(128) stage_apply(
    const FT* __restrict__ feature, const int* __restrict__ eidx,
    const unsigned short* __restrict__ wgt,
    const float* __restrict__ pa, const float* __restrict__ pc,
    float* __restrict__ out, float* __restrict__ cnt) {
  __shared__ __align__(16) _Float16 sW[WGT_HALFS];
  {
    const uint4* g = (const uint4*)wgt;
    uint4* s = (uint4*)sW;
    for (int i = threadIdx.x; i < WGT_HALFS / 8; i += blockDim.x) s[i] = g[i];
  }
  __syncthreads();

  const int lane   = threadIdx.x & 31;
  const int wave   = threadIdx.x >> 5;
  const int waveId = blockIdx.x * (blockDim.x >> 5) + wave;
  const int nWaves = gridDim.x * (blockDim.x >> 5);
  const int hi     = lane >> 4;

  float av[NT_TILES], cv[NT_TILES];
  #pragma unroll
  for (int nt = 0; nt < NT_TILES; ++nt) {
    const int col = nt * 16 + (lane & 15);
    av[nt] = pa[col];
    cv[nt] = pc[col];
  }

  v8f acc[NT_TILES];
  for (int t = waveId; t < TILES; t += nWaves) {
    const int didx = edge_tile_gemm(feature, eidx, sW, t, lane, acc);
    if (lane < 16 && (t * 16 + lane) < N_EDGES) atomicAdd(&cnt[didx], 1.0f);
    #pragma unroll
    for (int r = 0; r < 8; ++r) {
      const int rowLane = hi * 8 + r;                 // lane holding this row's dst
      const int node = __shfl(didx, rowLane, 32);
      const bool v = (t * 16 + rowLane) < N_EDGES;
      #pragma unroll
      for (int nt = 0; nt < NT_TILES; ++nt) {
        float y = fmaf(av[nt], acc[nt][r], cv[nt]);   // BN affine (bias folded)
        y = (y >= 0.0f) ? y : LRELU * y;              // LeakyReLU(0.3)
        if (v) atomicAdd(&out[(size_t)node * OUT_DIM + nt * 16 + (lane & 15)], y);
      }
    }
  }
}

__global__ void divide_kernel(float* __restrict__ out, const float* __restrict__ cnt) {
  const size_t stride = (size_t)gridDim.x * blockDim.x;
  for (size_t i = (size_t)blockIdx.x * blockDim.x + threadIdx.x;
       i < (size_t)N_NODES * OUT_DIM; i += stride) {
    out[i] = out[i] / fmaxf(cnt[i >> 6], 1.0f);
  }
}

// ---------------------------------------------------------------- launch

extern "C" void kernel_launch(void* const* d_in, const int* in_sizes, int n_in,
                              void* d_out, int out_size, void* d_ws, size_t ws_size,
                              hipStream_t stream) {
  const float* feature = (const float*)d_in[0];
  const int*   eidx    = (const int*)d_in[1];   // [2, E] flat: src then dst
  const float* W       = (const float*)d_in[2]; // [64, 128] row-major
  // d_in[3] (bias) unused: cancels through BatchNorm
  const float* gamma   = (const float*)d_in[4];
  const float* beta    = (const float*)d_in[5];
  float* out = (float*)d_out;

  const bool use_f16_feat =
      ws_size >= HFEAT_BYTES + WGT_BYTES + STATS_BYTES + CNT_BYTES;

  // Workspace layout:
  //   fast path: [hfeat 12.8MB][wgt 16KB][stats 1KB][cnt 400KB]
  //   fallback : [wgt 16KB][stats 1KB][cnt 400KB]
  char* base = (char*)d_ws;
  unsigned short* hfeat = (unsigned short*)base;
  unsigned short* wgt =
      (unsigned short*)(base + (use_f16_feat ? HFEAT_BYTES : 0));
  float* stats = (float*)((char*)wgt + WGT_BYTES);
  float* gsum = stats;
  float* gsq  = stats + 64;
  float* pa   = stats + 128;
  float* pc   = stats + 192;
  float* cnt  = stats + 256;

  zero_kernel<<<2048, 256, 0, stream>>>(out, cnt, stats);
  prep_weights<<<WGT_HALFS / 256, 256, 0, stream>>>(W, wgt);

  if (use_f16_feat) {
    const int nconv = (N_NODES * IN_DIM + 255) / 256;
    convert_features<<<nconv, 256, 0, stream>>>(feature, hfeat);
    stage_stats<unsigned short><<<1250, 128, 0, stream>>>(hfeat, eidx, wgt, gsum, gsq);
    finalize_params<<<1, 64, 0, stream>>>(gsum, gsq, gamma, beta, pa, pc);
    stage_apply<unsigned short><<<1250, 128, 0, stream>>>(hfeat, eidx, wgt, pa, pc, out, cnt);
  } else {
    stage_stats<float><<<1250, 128, 0, stream>>>(feature, eidx, wgt, gsum, gsq);
    finalize_params<<<1, 64, 0, stream>>>(gsum, gsq, gamma, beta, pa, pc);
    stage_apply<float><<<1250, 128, 0, stream>>>(feature, eidx, wgt, pa, pc, out, cnt);
  }

  divide_kernel<<<2048, 256, 0, stream>>>(out, cnt);
}
